// CounterfactualGenerator_74105365725292
// MI455X (gfx1250) — compile-verified, matrix operand order checked by reading, not run on hardware
//
#include <hip/hip_runtime.h>
#include <hip/hip_bf16.h>

#define BDIM 32768
#define VDIM 256

typedef __attribute__((ext_vector_type(8)))  float        v8f;
typedef __attribute__((ext_vector_type(8)))  unsigned int v8u;
typedef __attribute__((ext_vector_type(16))) __bf16       v16bf;
typedef int v4i32 __attribute__((vector_size(16)));

// ---------- async LDS staging support (safe fallback if builtin absent) ----------
#if defined(__has_builtin)
# if __has_builtin(__builtin_amdgcn_global_load_async_to_lds_b128)
#  define HAVE_ASYNC_LDS 1
# endif
#endif
#ifndef HAVE_ASYNC_LDS
# define HAVE_ASYNC_LDS 0
#endif

static __device__ __forceinline__ void stage16(void* lds_dst, const void* g_src) {
#if HAVE_ASYNC_LDS
  __builtin_amdgcn_global_load_async_to_lds_b128(
      (__attribute__((address_space(1))) v4i32*)(void*)g_src,
      (__attribute__((address_space(3))) v4i32*)lds_dst, 0, 0);
#else
  *(uint4*)lds_dst = *(const uint4*)g_src;
#endif
}
static __device__ __forceinline__ void stage_fence() {
#if HAVE_ASYNC_LDS
# if __has_builtin(__builtin_amdgcn_s_wait_asynccnt)
  __builtin_amdgcn_s_wait_asynccnt(0);
# else
  asm volatile("s_wait_asynccnt 0x0" ::: "memory");
# endif
#endif
  __syncthreads();
}

// ---------- bf16 helpers (round-to-nearest-even) ----------
static __device__ __forceinline__ unsigned short f2bf(float f) {
  unsigned u = __float_as_uint(f);
  u += 0x7FFFu + ((u >> 16) & 1u);
  return (unsigned short)(u >> 16);
}
static __device__ __forceinline__ unsigned pack2bf(float lo, float hi) {
  return (unsigned)f2bf(lo) | ((unsigned)f2bf(hi) << 16);
}
static __device__ __forceinline__ float bf2f(unsigned short b) {
  return __uint_as_float(((unsigned)b) << 16);
}

// ---------- WMMA wrapper ----------
static __device__ __forceinline__ v8f wmma_bf16(v8u a, v8u b, v8f c) {
  return __builtin_amdgcn_wmma_f32_16x16x32_bf16(
      false, __builtin_bit_cast(v16bf, a),
      false, __builtin_bit_cast(v16bf, b),
      (short)0, c, false, false);
}

// ---------- fragment loaders ----------
// 16x32 bf16 A-fragment from f32 row-major: lanes 0-15 M=lane K={0..7,16..23},
// lanes 16-31 M=lane-16 K={8..15,24..31}.  B is fed as B^T in the mirrored layout.
static __device__ __forceinline__ v8u load_frag_f32(const float* __restrict__ src, int ld,
                                                    int rowBase, int k0, int lane) {
  int hi = lane >> 4, l16 = lane & 15;
  const float* p = src + (size_t)(rowBase + l16) * ld + k0 + hi * 8;
  v8u u;
#pragma unroll
  for (int j = 0; j < 4; ++j) u[j] = pack2bf(p[2 * j], p[2 * j + 1]);
#pragma unroll
  for (int j = 0; j < 4; ++j) u[4 + j] = pack2bf(p[16 + 2 * j], p[16 + 2 * j + 1]);
  return u;
}
static __device__ __forceinline__ v8u load_frag_bf16(const unsigned short* __restrict__ src, int ld,
                                                     int n0, int k0, int lane) {
  int hi = lane >> 4, l16 = lane & 15;
  const unsigned short* p = src + (size_t)(n0 + l16) * ld + k0 + hi * 8;
  v8u u;
#pragma unroll
  for (int j = 0; j < 4; ++j) u[j] = *(const unsigned*)(p + 2 * j);
#pragma unroll
  for (int j = 0; j < 4; ++j) u[4 + j] = *(const unsigned*)(p + 16 + 2 * j);
  return u;
}
// A-fragment for the "intervened" segment of pin: x with cols 0..3 replaced by iv.
static __device__ __forceinline__ v8u load_frag_interv(const float* __restrict__ x,
                                                       const float* __restrict__ iv,
                                                       int rowBase, int k0, int lane) {
  int hi = lane >> 4, l16 = lane & 15;
  int row = rowBase + l16;
  const float* xr = x + (size_t)row * VDIM;
  const float* ir = iv + (size_t)row * 4;
  int cb = k0 + hi * 8;
  v8u u;
#pragma unroll
  for (int j = 0; j < 4; ++j) {
    int c0 = cb + 2 * j, c1 = c0 + 1;
    float f0 = (c0 < 4) ? ir[c0] : xr[c0];
    float f1 = (c1 < 4) ? ir[c1] : xr[c1];
    u[j] = pack2bf(f0, f1);
  }
#pragma unroll
  for (int j = 0; j < 4; ++j) {  // cols >= 16, never intervened
    int c0 = cb + 16 + 2 * j;
    u[4 + j] = pack2bf(xr[c0], xr[c0 + 1]);
  }
  return u;
}

// ---------- fragment-ready LDS staging ----------
// Chunk layout: per (nt) tile: 32 lanes x 48B slots (2x16B frag data + 16B pad -> 2-way
// bank conflicts max on b128 DS reads). One chunk = ntCount tiles of one 32-wide K slice.
#define STAGE_TILE 1536
static __device__ __forceinline__ void stage_chunk(const unsigned short* __restrict__ W, int ldW,
                                                   char* buf, int kk, int ntCount, int tid) {
  int total = ntCount * 64;  // (nt, lane, 2 pieces)
  for (int p = tid; p < total; p += 256) {
    int nt = p >> 6, rem = p & 63;
    int L = rem & 31, j = rem >> 5;
    const unsigned short* src =
        W + (size_t)(nt * 16 + (L & 15)) * ldW + kk * 32 + (L >> 4) * 8 + j * 16;
    stage16(buf + nt * STAGE_TILE + L * 48 + j * 16, src);
  }
}
static __device__ __forceinline__ v8u frag_from_stage(const char* buf, int nt, int lane) {
  const uint4* p = (const uint4*)(buf + nt * STAGE_TILE + lane * 48);
  uint4 q0 = p[0], q1 = p[1];
  v8u u;
  u[0] = q0.x; u[1] = q0.y; u[2] = q0.z; u[3] = q0.w;
  u[4] = q1.x; u[5] = q1.y; u[6] = q1.z; u[7] = q1.w;
  return u;
}

// ---------- weight prep: f32 -> bf16 (+ transpose for est_W2) ----------
__global__ void k_prep(const float* __restrict__ eW1, const float* __restrict__ eW2,
                       const float* __restrict__ pW1, const float* __restrict__ pW2,
                       unsigned short* __restrict__ W1b, unsigned short* __restrict__ W2bT,
                       unsigned short* __restrict__ PW1b, unsigned short* __restrict__ PW2b) {
  int i = blockIdx.x * blockDim.x + threadIdx.x;
  if (i < 65536) {
    W1b[i] = f2bf(eW1[i]);  // est_W1 [ko, v] row-major already = B^T
    int v = i >> 8, ko = i & 255, k = ko >> 6, o = ko & 63;
    W2bT[i] = f2bf(eW2[((k << 8) + v) * 64 + o]);  // W2bT[v, ko] = est_W2[k, v, o]
  }
  if (i < 98304) PW1b[i] = f2bf(pW1[i]);  // [128,768] row-major = B^T
  if (i < 8192)  PW2b[i] = f2bf(pW2[i]);  // [64,128]  row-major = B^T
}

// ---------- GEMM1: h' = relu(x @ W1^T + b1) * mag  ->  bf16 [B,256] ----------
__global__ void __launch_bounds__(256) k_gemm_h(
    const float* __restrict__ x, const float* __restrict__ iv,
    const unsigned short* __restrict__ W1b, const float* __restrict__ b1,
    unsigned short* __restrict__ hp) {
  __shared__ __align__(16) char smem[2 * 16 * STAGE_TILE];  // 48KB double buffer
  const int tid = threadIdx.x, lane = tid & 31, w = tid >> 5;
  const int hi = lane >> 4, l16 = lane & 15;
  const int rowBase = blockIdx.x * 128 + w * 16;

  stage_chunk(W1b, VDIM, smem, 0, 16, tid);
  stage_fence();

  v8f acc[16];
#pragma unroll
  for (int nt = 0; nt < 16; ++nt) acc[nt] = (v8f){0.f, 0.f, 0.f, 0.f, 0.f, 0.f, 0.f, 0.f};

  for (int kk = 0; kk < 8; ++kk) {
    const char* cur = smem + (kk & 1) * (16 * STAGE_TILE);
    if (kk < 7) stage_chunk(W1b, VDIM, smem + ((kk + 1) & 1) * (16 * STAGE_TILE), kk + 1, 16, tid);
    v8u a = load_frag_f32(x, VDIM, rowBase, kk * 32, lane);
#pragma unroll
    for (int nt = 0; nt < 16; ++nt) acc[nt] = wmma_bf16(a, frag_from_stage(cur, nt, lane), acc[nt]);
    stage_fence();
  }

  float mg[4][8];
#pragma unroll
  for (int r = 0; r < 8; ++r) {
    int row = rowBase + r + hi * 8;
    const float* xr = x + (size_t)row * VDIM;
    const float* ir = iv + (size_t)row * 4;
#pragma unroll
    for (int k = 0; k < 4; ++k) mg[k][r] = fabsf(ir[k] - xr[k]);
  }
#pragma unroll
  for (int nt = 0; nt < 16; ++nt) {
    float bias = b1[nt * 16 + l16];
#pragma unroll
    for (int r = 0; r < 8; ++r) {
      int m = r + hi * 8;
      float v = acc[nt][r] + bias;
      v = v > 0.f ? v : 0.f;
      v *= mg[nt >> 2][r];
      hp[(size_t)(rowBase + m) * VDIM + nt * 16 + l16] = f2bf(v);
    }
  }
}

// ---------- GEMM2: effects = h' @ W2bT^T (+ mag.b2); data0 = intervened + effects ----------
__global__ void __launch_bounds__(256) k_gemm_eff(
    const float* __restrict__ x, const float* __restrict__ iv,
    const unsigned short* __restrict__ hp, const unsigned short* __restrict__ W2bT,
    const float* __restrict__ b2, float* __restrict__ effOut, float* __restrict__ data0) {
  __shared__ __align__(16) char smem[2 * 16 * STAGE_TILE];
  const int tid = threadIdx.x, lane = tid & 31, w = tid >> 5;
  const int hi = lane >> 4, l16 = lane & 15;
  const int rowBase = blockIdx.x * 128 + w * 16;

  stage_chunk(W2bT, VDIM, smem, 0, 16, tid);
  stage_fence();

  v8f acc[16];
#pragma unroll
  for (int nt = 0; nt < 16; ++nt) acc[nt] = (v8f){0.f, 0.f, 0.f, 0.f, 0.f, 0.f, 0.f, 0.f};

  for (int kk = 0; kk < 8; ++kk) {
    const char* cur = smem + (kk & 1) * (16 * STAGE_TILE);
    if (kk < 7) stage_chunk(W2bT, VDIM, smem + ((kk + 1) & 1) * (16 * STAGE_TILE), kk + 1, 16, tid);
    v8u a = load_frag_bf16(hp, VDIM, rowBase, kk * 32, lane);
#pragma unroll
    for (int nt = 0; nt < 16; ++nt) acc[nt] = wmma_bf16(a, frag_from_stage(cur, nt, lane), acc[nt]);
    stage_fence();
  }

  float mg[4][8];
#pragma unroll
  for (int r = 0; r < 8; ++r) {
    int row = rowBase + r + hi * 8;
    const float* xr = x + (size_t)row * VDIM;
    const float* ir = iv + (size_t)row * 4;
#pragma unroll
    for (int k = 0; k < 4; ++k) mg[k][r] = fabsf(ir[k] - xr[k]);
  }
#pragma unroll
  for (int nt = 0; nt < 16; ++nt) {
    int v = nt * 16 + l16;
    float b2v[4];
#pragma unroll
    for (int k = 0; k < 4; ++k) b2v[k] = b2[k * VDIM + v];
#pragma unroll
    for (int r = 0; r < 8; ++r) {
      int row = rowBase + r + hi * 8;
      float eff = acc[nt][r];
#pragma unroll
      for (int k = 0; k < 4; ++k) eff += mg[k][r] * b2v[k];
      float base = (v < 4) ? iv[(size_t)row * 4 + v] : x[(size_t)row * VDIM + v];
      effOut[(size_t)row * VDIM + v] = eff;
      data0[(size_t)row * VDIM + v] = base + eff;
    }
  }
}

// ---------- propagation partial products: Q_g = A_{g*256+255}...A_{g*256} (f32, LDS) ----------
__global__ void __launch_bounds__(64) k_partials(
    const int* __restrict__ cause, const int* __restrict__ effect,
    const float* __restrict__ strength, float* __restrict__ Q) {
  __shared__ float Mrow[256 * 64];  // word addr = row*64 + lane -> conflict-free
  int t = threadIdx.x;
  int g = blockIdx.x >> 2, cg = blockIdx.x & 3;
  int C = cg * 64 + t;  // global column owned by this thread
  for (int r = 0; r < 256; ++r) Mrow[r * 64 + t] = (r == C) ? 1.f : 0.f;
  int base = g * 256;
  for (int i = 0; i < 256; ++i) {  // uniform edge reads -> scalar loads
    int c = cause[base + i], e = effect[base + i];
    float s = strength[base + i] * 0.1f;
    Mrow[e * 64 + t] += s * Mrow[c * 64 + t];  // columns independent: no barriers
  }
  for (int r = 0; r < 256; ++r) Q[(size_t)g * 65536 + r * 256 + C] = Mrow[r * 64 + t];
}

// ---------- small f32 256x256x256 GEMM: D = A*B (optionally emit bf16(D - I)) ----------
__global__ void __launch_bounds__(256) k_matmul256(
    const float* __restrict__ A, const float* __restrict__ Bm,
    float* __restrict__ D, unsigned short* __restrict__ deltaOut) {
  int i = blockIdx.x, j = threadIdx.x;
  float s = 0.f;
  for (int k = 0; k < 256; ++k) s = fmaf(A[i * 256 + k], Bm[k * 256 + j], s);
  D[i * 256 + j] = s;
  if (deltaOut) deltaOut[i * 256 + j] = f2bf(s - (i == j ? 1.f : 0.f));
}

// ---------- apply M: final_cf = data0 + data0 @ (M-I)^T; fused impact = ||cf - x|| ----------
__global__ void __launch_bounds__(256) k_applyM(
    const float* __restrict__ d0, const unsigned short* __restrict__ Mb,
    const float* __restrict__ x, float* __restrict__ cf, float* __restrict__ impact) {
  __shared__ __align__(16) char smem[2 * 16 * STAGE_TILE];
  const int tid = threadIdx.x, lane = tid & 31, w = tid >> 5;
  const int hi = lane >> 4, l16 = lane & 15;
  const int rowBase = blockIdx.x * 128 + w * 16;

  stage_chunk(Mb, VDIM, smem, 0, 16, tid);
  stage_fence();

  v8f acc[16];
#pragma unroll
  for (int nt = 0; nt < 16; ++nt) acc[nt] = (v8f){0.f, 0.f, 0.f, 0.f, 0.f, 0.f, 0.f, 0.f};

  for (int kk = 0; kk < 8; ++kk) {
    const char* cur = smem + (kk & 1) * (16 * STAGE_TILE);
    if (kk < 7) stage_chunk(Mb, VDIM, smem + ((kk + 1) & 1) * (16 * STAGE_TILE), kk + 1, 16, tid);
    v8u a = load_frag_f32(d0, VDIM, rowBase, kk * 32, lane);
#pragma unroll
    for (int nt = 0; nt < 16; ++nt) acc[nt] = wmma_bf16(a, frag_from_stage(cur, nt, lane), acc[nt]);
    stage_fence();
  }

  float rs[8];
#pragma unroll
  for (int r = 0; r < 8; ++r) rs[r] = 0.f;
#pragma unroll
  for (int nt = 0; nt < 16; ++nt) {
    int v = nt * 16 + l16;
#pragma unroll
    for (int r = 0; r < 8; ++r) {
      int row = rowBase + r + hi * 8;
      float cfv = acc[nt][r] + d0[(size_t)row * VDIM + v];
      float dx = cfv - x[(size_t)row * VDIM + v];
      rs[r] += dx * dx;
      cf[(size_t)row * VDIM + v] = cfv;
    }
  }
#pragma unroll
  for (int r = 0; r < 8; ++r) {
#pragma unroll
    for (int mask = 1; mask < 16; mask <<= 1) rs[r] += __shfl_xor(rs[r], mask, 32);
  }
  if (l16 == 0) {
#pragma unroll
    for (int r = 0; r < 8; ++r) impact[rowBase + r + hi * 8] = sqrtf(rs[r]);
  }
}

// ---------- plausibility MLP: [B,768]->128 relu ->64 relu ->1 sigmoid ----------
__global__ void __launch_bounds__(256) k_plaus(
    const float* __restrict__ x, const float* __restrict__ iv, const float* __restrict__ cf,
    const unsigned short* __restrict__ PW1b, const float* __restrict__ pb1,
    const unsigned short* __restrict__ PW2b, const float* __restrict__ pb2,
    const float* __restrict__ pW3, const float* __restrict__ pb3,
    float* __restrict__ plausOut) {
  __shared__ __align__(16) char smem[32768 + 2 * 8 * STAGE_TILE];  // p1s | stage/p2s (reused)
  unsigned short* p1s = (unsigned short*)smem;
  char* stg = smem + 32768;
  unsigned short* p2s = (unsigned short*)(smem + 32768);  // reused after phase 1
  const int tid = threadIdx.x, lane = tid & 31, w = tid >> 5;
  const int hi = lane >> 4, l16 = lane & 15;
  const int rowBase = blockIdx.x * 128 + w * 16;

  stage_chunk(PW1b, 768, stg, 0, 8, tid);
  stage_fence();

  v8f acc[8];
#pragma unroll
  for (int nt = 0; nt < 8; ++nt) acc[nt] = (v8f){0.f, 0.f, 0.f, 0.f, 0.f, 0.f, 0.f, 0.f};

  for (int kk = 0; kk < 24; ++kk) {  // pin = [x | intervened | final_cf]
    const char* cur = stg + (kk & 1) * (8 * STAGE_TILE);
    if (kk < 23) stage_chunk(PW1b, 768, stg + ((kk + 1) & 1) * (8 * STAGE_TILE), kk + 1, 8, tid);
    v8u a;
    if (kk < 8)       a = load_frag_f32(x, VDIM, rowBase, kk * 32, lane);
    else if (kk < 16) a = load_frag_interv(x, iv, rowBase, (kk - 8) * 32, lane);
    else              a = load_frag_f32(cf, VDIM, rowBase, (kk - 16) * 32, lane);
#pragma unroll
    for (int nt = 0; nt < 8; ++nt) acc[nt] = wmma_bf16(a, frag_from_stage(cur, nt, lane), acc[nt]);
    stage_fence();
  }
#pragma unroll
  for (int nt = 0; nt < 8; ++nt) {
    float bias = pb1[nt * 16 + l16];
#pragma unroll
    for (int r = 0; r < 8; ++r) {
      int m = r + hi * 8;
      float v = acc[nt][r] + bias;
      p1s[w * 2048 + m * 128 + nt * 16 + l16] = f2bf(v > 0.f ? v : 0.f);
    }
  }
  __syncthreads();  // p1s ready; stage region now dead -> reuse as p2s

  v8f acc2[4];
#pragma unroll
  for (int nt = 0; nt < 4; ++nt) acc2[nt] = (v8f){0.f, 0.f, 0.f, 0.f, 0.f, 0.f, 0.f, 0.f};
  for (int kk = 0; kk < 4; ++kk) {
    v8u a = load_frag_bf16(p1s + w * 2048, 128, 0, kk * 32, lane);
#pragma unroll
    for (int nt = 0; nt < 4; ++nt) {
      v8u b = load_frag_bf16(PW2b, 128, nt * 16, kk * 32, lane);
      acc2[nt] = wmma_bf16(a, b, acc2[nt]);
    }
  }
#pragma unroll
  for (int nt = 0; nt < 4; ++nt) {
    float bias = pb2[nt * 16 + l16];
#pragma unroll
    for (int r = 0; r < 8; ++r) {
      int m = r + hi * 8;
      float v = acc2[nt][r] + bias;
      p2s[w * 1024 + m * 64 + nt * 16 + l16] = f2bf(v > 0.f ? v : 0.f);
    }
  }
  __syncthreads();

  if (lane < 16) {  // lane handles one of the wave's 16 rows
    float s = pb3[0];
    for (int o = 0; o < 64; ++o) s += bf2f(p2s[w * 1024 + lane * 64 + o]) * pW3[o];
    plausOut[rowBase + lane] = 1.f / (1.f + expf(-s));
  }
}

extern "C" void kernel_launch(void* const* d_in, const int* in_sizes, int n_in,
                              void* d_out, int out_size, void* d_ws, size_t ws_size,
                              hipStream_t stream) {
  (void)in_sizes; (void)n_in; (void)out_size; (void)ws_size;
  const float* x    = (const float*)d_in[0];
  const float* iv   = (const float*)d_in[1];
  const float* str  = (const float*)d_in[2];
  // d_in[3] = intervention_variables == arange(4)
  const int*   cause  = (const int*)d_in[4];
  const int*   effect = (const int*)d_in[5];
  const float* eW1 = (const float*)d_in[6];
  const float* eb1 = (const float*)d_in[7];
  const float* eW2 = (const float*)d_in[8];
  const float* eb2 = (const float*)d_in[9];
  const float* pW1 = (const float*)d_in[10];
  const float* pb1 = (const float*)d_in[11];
  const float* pW2 = (const float*)d_in[12];
  const float* pb2 = (const float*)d_in[13];
  const float* pW3 = (const float*)d_in[14];
  const float* pb3 = (const float*)d_in[15];

  float* out    = (float*)d_out;
  float* cf     = out;                                   // [B,V]
  float* effOut = out + (size_t)BDIM * VDIM;             // [B,V]
  float* plausO = out + 2 * (size_t)BDIM * VDIM;         // [B,1]
  float* impact = plausO + BDIM;                         // [B,1]

  char* ws = (char*)d_ws;
  unsigned short* W1b  = (unsigned short*)(ws + 0);         // 256x256 bf16
  unsigned short* W2bT = (unsigned short*)(ws + 131072);    // 256x256 bf16
  unsigned short* PW1b = (unsigned short*)(ws + 262144);    // 128x768 bf16
  unsigned short* PW2b = (unsigned short*)(ws + 458752);    // 64x128 bf16
  unsigned short* hp   = (unsigned short*)(ws + 475136);    // [B,256] bf16
  float* data0 = (float*)(ws + 17252352);                   // [B,256] f32
  float* Q     = (float*)(ws + 50806784);                   // 8 x 256x256 f32
  float* S     = (float*)(ws + 52903936);                   // 4 x 256x256 f32
  float* T     = (float*)(ws + 53952512);                   // 2 x 256x256 f32
  float* Pm    = (float*)(ws + 54476800);                   // 256x256 f32
  float* U     = (float*)(ws + 54738944);                   // 256x256 f32
  unsigned short* Mb = (unsigned short*)(ws + 55001088);    // bf16(M - I)

  k_prep<<<384, 256, 0, stream>>>(eW1, eW2, pW1, pW2, W1b, W2bT, PW1b, PW2b);
  k_partials<<<32, 64, 0, stream>>>(cause, effect, str, Q);
  k_gemm_h<<<256, 256, 0, stream>>>(x, iv, W1b, eb1, hp);
  k_gemm_eff<<<256, 256, 0, stream>>>(x, iv, hp, W2bT, eb2, effOut, data0);
  // P = Q7 Q6 Q5 Q4 Q3 Q2 Q1 Q0 via tree; then M = P^3.
  k_matmul256<<<256, 256, 0, stream>>>(Q + 1 * 65536, Q + 0 * 65536, S + 0 * 65536, nullptr);
  k_matmul256<<<256, 256, 0, stream>>>(Q + 3 * 65536, Q + 2 * 65536, S + 1 * 65536, nullptr);
  k_matmul256<<<256, 256, 0, stream>>>(Q + 5 * 65536, Q + 4 * 65536, S + 2 * 65536, nullptr);
  k_matmul256<<<256, 256, 0, stream>>>(Q + 7 * 65536, Q + 6 * 65536, S + 3 * 65536, nullptr);
  k_matmul256<<<256, 256, 0, stream>>>(S + 1 * 65536, S + 0 * 65536, T + 0 * 65536, nullptr);
  k_matmul256<<<256, 256, 0, stream>>>(S + 3 * 65536, S + 2 * 65536, T + 1 * 65536, nullptr);
  k_matmul256<<<256, 256, 0, stream>>>(T + 1 * 65536, T + 0 * 65536, Pm, nullptr);
  k_matmul256<<<256, 256, 0, stream>>>(Pm, Pm, U, nullptr);
  k_matmul256<<<256, 256, 0, stream>>>(U, Pm, S /*scratch for M*/, Mb);  // M=U*P, emit bf16(M-I)
  k_applyM<<<256, 256, 0, stream>>>(data0, Mb, x, cf, impact);
  k_plaus<<<256, 256, 0, stream>>>(x, iv, cf, PW1b, pb1, PW2b, pb2, pW3, pb3, plausO);
}
